// EquivariantConv_1864015807169
// MI455X (gfx1250) — compile-verified, matrix operand order checked by reading, not run on hardware
//
#include <hip/hip_runtime.h>
#include <cstdint>

// ---------------------------------------------------------------------------
// EGNN layer for MI455X (gfx1250, wave32, WMMA f32<=f16 16x16x32)
//   h_new, x_new = EquivariantConv(h, x, edge_index, ...)
// GEMMs on v_wmma_f32_16x16x32_f16; weights pre-swizzled to fragment-major
// layout so every B fragment is a single contiguous 32-byte load; LDS
// activation tiles stored in A-fragment layout (2x ds_load_b128 per frag).
// Segment sums via hardware global_atomic_add_f32.
// ---------------------------------------------------------------------------

typedef __attribute__((ext_vector_type(16))) _Float16 v16h;
typedef __attribute__((ext_vector_type(8)))  float    v8f;

#define HD  128      // hidden size H
#define KT1 9        // K-tiles for msg_w1 (2H+1 = 257 -> 288 = 9*32)

// K offset inside a 16x32 f16 A/B fragment for vgpr-pair v (0..7), half hh.
__device__ __host__ __forceinline__ int kmap(int v, int hh) {
    return (v < 4 ? 2 * v : 16 + 2 * (v - 4)) + hh * 8;
}

__device__ __forceinline__ v8f wmma_f16(v16h a, v16h b, v8f c) {
    return __builtin_amdgcn_wmma_f32_16x16x32_f16(
        /*neg_a=*/false, a, /*neg_b=*/false, b,
        /*c_mod=*/(short)0, c, /*reuse_a=*/false, /*reuse_b=*/false);
}

// B fragment: contiguous 32 bytes at [((kt*8+nt)*32+lane)*16]
__device__ __forceinline__ v16h load_b_frag(const _Float16* __restrict__ wp,
                                            int kt, int nt, int lane) {
    const _Float16* p = wp + (size_t)(((kt * 8 + nt) * 32 + lane) << 4);
    return *(const v16h*)__builtin_assume_aligned(p, 32);
}

// A fragment from LDS tile stored in fragment layout: [kt][lane][16]
__device__ __forceinline__ v16h load_a_frag_lds(const _Float16* mt,
                                                int kt, int lane) {
    const _Float16* p = mt + (((kt * 32) + lane) << 4);
    return *(const v16h*)__builtin_assume_aligned(p, 32);
}

// where value (row = hh*8+v, chan n = nt*16+m) of a 16x128 tile lives in
// A-fragment layout (inverse of kmap; folds to cheap bit math)
__device__ __forceinline__ int afrag_addr(int nt, int v, int m, int hh) {
    int kt    = nt >> 1;
    int lane2 = ((m >> 3) << 4) + hh * 8 + v;
    int e     = (m & 7) + ((nt & 1) << 3);
    return ((kt * 32 + lane2) << 4) + e;
}

__device__ __forceinline__ float silu_f(float y) {
    return y / (1.f + __expf(-y));
}

__device__ __forceinline__ v8f v8zero() {
    v8f z = {0.f, 0.f, 0.f, 0.f, 0.f, 0.f, 0.f, 0.f};
    return z;
}

// ---------------------------------------------------------------------------
__global__ void zero_kernel(float* __restrict__ p, size_t n) {
    size_t i = (size_t)blockIdx.x * blockDim.x + threadIdx.x;
    if (i < n) p[i] = 0.f;
}

// f32 weight [Ksrc][128] -> f16 fragment-major [KT][8][32][16] (zero padded)
__global__ void convert_w_frag_kernel(const float* __restrict__ src,
                                      _Float16* __restrict__ dst,
                                      int Ksrc, int KT) {
    int idx = blockIdx.x * blockDim.x + threadIdx.x;
    int total = KT * 8 * 32 * 16;
    if (idx >= total) return;
    int e    = idx & 15;
    int lane = (idx >> 4) & 31;
    int nt   = (idx >> 9) & 7;
    int kt   = idx >> 12;
    int v = e >> 1, s = e & 1, hh = lane >> 4;
    int k = kt * 32 + kmap(v, hh) + s;
    int n = nt * 16 + (lane & 15);
    float val = (k < Ksrc) ? src[(size_t)k * HD + n] : 0.f;
    dst[idx] = (_Float16)val;
}

// ---------------------------------------------------------------------------
// Edge kernel: one wave handles a tile of 16 edges.
#define EW 4
__global__ __launch_bounds__(EW * 32) void egnn_edge_kernel(
    const float* __restrict__ h, const float* __restrict__ x,
    const long long* __restrict__ ei,
    const _Float16* __restrict__ w1p, const float* __restrict__ b1,
    const _Float16* __restrict__ w2p, const float* __restrict__ b2,
    const _Float16* __restrict__ cw1p, const float* __restrict__ cb1,
    const float* __restrict__ cw2,
    float* __restrict__ agg, float* __restrict__ coordacc,
    float* __restrict__ deg, long long Ee)
{
    __shared__ __align__(64) _Float16 mlds[EW][16 * HD];
    const int lane = threadIdx.x & 31;
    const int wid  = threadIdx.x >> 5;
    const long long base = ((long long)blockIdx.x * EW + wid) * 16;
    if (base >= Ee) return;
    const int m = lane & 15, hh = lane >> 4;

    long long em = base + m; if (em >= Ee) em = Ee - 1;
    const long long r = ei[em], c = ei[Ee + em];
    const float* __restrict__ hr = h + (size_t)r * HD;
    const float* __restrict__ hc = h + (size_t)c * HD;
    const float rx = x[r * 3 + 0] - x[c * 3 + 0];
    const float ry = x[r * 3 + 1] - x[c * 3 + 1];
    const float rz = x[r * 3 + 2] - x[c * 3 + 2];
    const float dist = sqrtf(rx * rx + ry * ry + rz * rz);
    _Float16* mt = mlds[wid];

    // ---- GEMM1: mi[16 x 288] @ msg_w1[288 x 128], A gathered from global h
    v8f acc[8];
#pragma unroll
    for (int nt = 0; nt < 8; ++nt) acc[nt] = v8zero();
#pragma unroll
    for (int kt = 0; kt < KT1; ++kt) {
        v16h a;
#pragma unroll
        for (int v = 0; v < 8; ++v) {
            int kb = kt * 32 + kmap(v, hh);
#pragma unroll
            for (int s = 0; s < 2; ++s) {
                int k = kb + s;
                float val;
                if (k < HD)            val = hr[k];
                else if (k < 2 * HD)   val = hc[k - HD];
                else if (k == 2 * HD)  val = dist;
                else                   val = 0.f;
                a[2 * v + s] = (_Float16)val;
            }
        }
#pragma unroll
        for (int nt = 0; nt < 8; ++nt)
            acc[nt] = wmma_f16(a, load_b_frag(w1p, kt, nt, lane), acc[nt]);
    }
    // bias + SiLU -> LDS in A-fragment layout
#pragma unroll
    for (int nt = 0; nt < 8; ++nt) {
        const float bias = b1[nt * 16 + m];
#pragma unroll
        for (int v = 0; v < 8; ++v) {
            float s = silu_f(acc[nt][v] + bias);
            mt[afrag_addr(nt, v, m, hh)] = (_Float16)s;
        }
    }
    asm volatile("s_wait_dscnt 0" ::: "memory");   // wave-local LDS RAW

    // ---- GEMM2: m1[16x128] @ msg_w2[128x128]
    v8f acc2[8];
#pragma unroll
    for (int nt = 0; nt < 8; ++nt) acc2[nt] = v8zero();
#pragma unroll
    for (int kt = 0; kt < 4; ++kt) {
        v16h a = load_a_frag_lds(mt, kt, lane);
#pragma unroll
        for (int nt = 0; nt < 8; ++nt)
            acc2[nt] = wmma_f16(a, load_b_frag(w2p, kt, nt, lane), acc2[nt]);
    }
    // col index + validity for the 8 edge rows of this lane's C fragments
    long long ce[8]; bool ev[8];
#pragma unroll
    for (int v = 0; v < 8; ++v) {
        long long er = base + hh * 8 + v;
        ev[v] = er < Ee;
        ce[v] = ei[Ee + (ev[v] ? er : (Ee - 1))];
    }
    // bias + SiLU -> (a) atomic scatter into agg, (b) LDS for coord head
#pragma unroll
    for (int nt = 0; nt < 8; ++nt) {
        const float bias = b2[nt * 16 + m];
#pragma unroll
        for (int v = 0; v < 8; ++v) {
            float s = silu_f(acc2[nt][v] + bias);
            mt[afrag_addr(nt, v, m, hh)] = (_Float16)s;
            if (ev[v])
                unsafeAtomicAdd(&agg[ce[v] * HD + nt * 16 + m], s);
        }
    }
    asm volatile("s_wait_dscnt 0" ::: "memory");

    // ---- coord head: silu(m @ coord_w1 + b) . coord_w2 -> tanh
    v8f acc3[8];
#pragma unroll
    for (int nt = 0; nt < 8; ++nt) acc3[nt] = v8zero();
#pragma unroll
    for (int kt = 0; kt < 4; ++kt) {
        v16h a = load_a_frag_lds(mt, kt, lane);
#pragma unroll
        for (int nt = 0; nt < 8; ++nt)
            acc3[nt] = wmma_f16(a, load_b_frag(cw1p, kt, nt, lane), acc3[nt]);
    }
    float pd[8] = {0.f, 0.f, 0.f, 0.f, 0.f, 0.f, 0.f, 0.f};
#pragma unroll
    for (int nt = 0; nt < 8; ++nt) {
        const float bias = cb1[nt * 16 + m];
        const float w2n  = cw2[nt * 16 + m];
#pragma unroll
        for (int v = 0; v < 8; ++v)
            pd[v] += silu_f(acc3[nt][v] + bias) * w2n;
    }
#pragma unroll
    for (int mk = 1; mk < 16; mk <<= 1) {
#pragma unroll
        for (int v = 0; v < 8; ++v)
            pd[v] += __shfl_xor(pd[v], mk, 32);
    }
    // one lane per edge commits the coordinate update
#pragma unroll
    for (int v = 0; v < 8; ++v) {
        if (m == v && ev[v]) {
            long long er = base + hh * 8 + v;
            long long rr = ei[er], cc = ei[Ee + er];
            float ax = x[rr * 3 + 0] - x[cc * 3 + 0];
            float ay = x[rr * 3 + 1] - x[cc * 3 + 1];
            float az = x[rr * 3 + 2] - x[cc * 3 + 2];
            float d2 = sqrtf(ax * ax + ay * ay + az * az);
            float sc = tanhf(pd[v]) / (d2 + 1e-8f);
            unsafeAtomicAdd(&coordacc[cc * 3 + 0], sc * ax);
            unsafeAtomicAdd(&coordacc[cc * 3 + 1], sc * ay);
            unsafeAtomicAdd(&coordacc[cc * 3 + 2], sc * az);
            unsafeAtomicAdd(&deg[cc], 1.f);
        }
    }
}

// ---------------------------------------------------------------------------
// Node kernel: one wave handles a tile of 16 nodes; fused MLP + LayerNorm.
#define NW 4
__global__ __launch_bounds__(NW * 32) void egnn_node_kernel(
    const float* __restrict__ h, const float* __restrict__ agg,
    const _Float16* __restrict__ nw1p, const float* __restrict__ nb1,
    const _Float16* __restrict__ nw2p, const float* __restrict__ nb2,
    const float* __restrict__ ln_g, const float* __restrict__ ln_b,
    float* __restrict__ hout, int Nn)
{
    __shared__ __align__(64) _Float16 ulds[NW][16 * HD];
    const int lane = threadIdx.x & 31;
    const int wid  = threadIdx.x >> 5;
    const long long base = ((long long)blockIdx.x * NW + wid) * 16;
    if (base >= Nn) return;
    const int m = lane & 15, hh = lane >> 4;
    long long nm = base + m; if (nm >= Nn) nm = Nn - 1;
    const float* __restrict__ hp = h   + (size_t)nm * HD;
    const float* __restrict__ ap = agg + (size_t)nm * HD;
    _Float16* ut = ulds[wid];

    // ---- GEMM1: [h | agg] [16 x 256] @ node_w1[256 x 128]
    v8f acc[8];
#pragma unroll
    for (int nt = 0; nt < 8; ++nt) acc[nt] = v8zero();
#pragma unroll
    for (int kt = 0; kt < 8; ++kt) {
        v16h a;
#pragma unroll
        for (int v = 0; v < 8; ++v) {
            int kb = kt * 32 + kmap(v, hh);
#pragma unroll
            for (int s = 0; s < 2; ++s) {
                int k = kb + s;
                float val = (k < HD) ? hp[k] : ap[k - HD];
                a[2 * v + s] = (_Float16)val;
            }
        }
#pragma unroll
        for (int nt = 0; nt < 8; ++nt)
            acc[nt] = wmma_f16(a, load_b_frag(nw1p, kt, nt, lane), acc[nt]);
    }
#pragma unroll
    for (int nt = 0; nt < 8; ++nt) {
        const float bias = nb1[nt * 16 + m];
#pragma unroll
        for (int v = 0; v < 8; ++v) {
            float s = silu_f(acc[nt][v] + bias);
            ut[afrag_addr(nt, v, m, hh)] = (_Float16)s;
        }
    }
    asm volatile("s_wait_dscnt 0" ::: "memory");

    // ---- GEMM2: u[16x128] @ node_w2[128x128]
    v8f acc2[8];
#pragma unroll
    for (int nt = 0; nt < 8; ++nt) acc2[nt] = v8zero();
#pragma unroll
    for (int kt = 0; kt < 4; ++kt) {
        v16h a = load_a_frag_lds(ut, kt, lane);
#pragma unroll
        for (int nt = 0; nt < 8; ++nt)
            acc2[nt] = wmma_f16(a, load_b_frag(nw2p, kt, nt, lane), acc2[nt]);
    }
    // residual + bias, then LayerNorm across the 128 channels of each row
    float fin[8][8];
    float ps[8] = {0.f, 0.f, 0.f, 0.f, 0.f, 0.f, 0.f, 0.f};
    float pq[8] = {0.f, 0.f, 0.f, 0.f, 0.f, 0.f, 0.f, 0.f};
#pragma unroll
    for (int nt = 0; nt < 8; ++nt) {
        const float bias = nb2[nt * 16 + m];
#pragma unroll
        for (int v = 0; v < 8; ++v) {
            long long row = base + hh * 8 + v;
            long long rc  = row < Nn ? row : (Nn - 1);
            float val = acc2[nt][v] + bias + h[(size_t)rc * HD + nt * 16 + m];
            fin[nt][v] = val;
            ps[v] += val;
            pq[v] += val * val;
        }
    }
#pragma unroll
    for (int mk = 1; mk < 16; mk <<= 1) {
#pragma unroll
        for (int v = 0; v < 8; ++v) {
            ps[v] += __shfl_xor(ps[v], mk, 32);
            pq[v] += __shfl_xor(pq[v], mk, 32);
        }
    }
    float mu[8], rs[8];
#pragma unroll
    for (int v = 0; v < 8; ++v) {
        mu[v] = ps[v] * (1.f / HD);
        float var = pq[v] * (1.f / HD) - mu[v] * mu[v];
        rs[v] = rsqrtf(var + 1e-5f);
    }
#pragma unroll
    for (int nt = 0; nt < 8; ++nt) {
        const float g  = ln_g[nt * 16 + m];
        const float bb = ln_b[nt * 16 + m];
#pragma unroll
        for (int v = 0; v < 8; ++v) {
            long long row = base + hh * 8 + v;
            if (row < Nn)
                hout[(size_t)row * HD + nt * 16 + m] =
                    (fin[nt][v] - mu[v]) * rs[v] * g + bb;
        }
    }
}

// ---------------------------------------------------------------------------
// x_new = x + coord_update / (deg + 1)
__global__ void xnew_kernel(const float* __restrict__ x,
                            const float* __restrict__ coordacc,
                            const float* __restrict__ deg,
                            float* __restrict__ outx, int Nn) {
    int i = blockIdx.x * blockDim.x + threadIdx.x;
    if (i >= Nn) return;
    float d = deg[i] + 1.f;
#pragma unroll
    for (int t = 0; t < 3; ++t)
        outx[(size_t)i * 3 + t] = x[(size_t)i * 3 + t] + coordacc[(size_t)i * 3 + t] / d;
}

// ---------------------------------------------------------------------------
extern "C" void kernel_launch(void* const* d_in, const int* in_sizes, int n_in,
                              void* d_out, int out_size, void* d_ws, size_t ws_size,
                              hipStream_t stream) {
    const float*     h        = (const float*)d_in[0];
    const float*     x        = (const float*)d_in[1];
    const long long* ei       = (const long long*)d_in[2];   // int64 edge_index
    const float*     msg_w1   = (const float*)d_in[3];
    const float*     msg_b1   = (const float*)d_in[4];
    const float*     msg_w2   = (const float*)d_in[5];
    const float*     msg_b2   = (const float*)d_in[6];
    const float*     coord_w1 = (const float*)d_in[7];
    const float*     coord_b1 = (const float*)d_in[8];
    const float*     coord_w2 = (const float*)d_in[9];
    const float*     node_w1  = (const float*)d_in[10];
    const float*     node_b1  = (const float*)d_in[11];
    const float*     node_w2  = (const float*)d_in[12];
    const float*     node_b2  = (const float*)d_in[13];
    const float*     ln_g     = (const float*)d_in[14];
    const float*     ln_b     = (const float*)d_in[15];

    const int       Nn = in_sizes[0] / HD;
    const long long Ee = (long long)(in_sizes[2] / 2);

    // workspace layout (~26.7 MB)
    float* agg      = (float*)d_ws;                 // N*128
    float* coordacc = agg + (size_t)Nn * HD;        // N*3
    float* deg      = coordacc + (size_t)Nn * 3;    // N
    uintptr_t wb = ((uintptr_t)(deg + Nn) + 63) & ~(uintptr_t)63;
    _Float16* w1p  = (_Float16*)wb;                 // 9*4096 halfs (frag-major)
    _Float16* w2p  = w1p  + (size_t)KT1 * 4096;     // 4*4096
    _Float16* cw1p = w2p  + (size_t)4 * 4096;       // 4*4096
    _Float16* nw1p = cw1p + (size_t)4 * 4096;       // 8*4096
    _Float16* nw2p = nw1p + (size_t)8 * 4096;       // 4*4096

    // zero the accumulators
    size_t zn = (size_t)Nn * HD + (size_t)Nn * 3 + (size_t)Nn;
    zero_kernel<<<(unsigned)((zn + 255) / 256), 256, 0, stream>>>(agg, zn);

    // weights -> f16 fragment-major
    convert_w_frag_kernel<<<(KT1 * 4096 + 255) / 256, 256, 0, stream>>>(msg_w1, w1p, 2 * HD + 1, KT1);
    convert_w_frag_kernel<<<(4 * 4096 + 255) / 256, 256, 0, stream>>>(msg_w2, w2p, HD, 4);
    convert_w_frag_kernel<<<(4 * 4096 + 255) / 256, 256, 0, stream>>>(coord_w1, cw1p, HD, 4);
    convert_w_frag_kernel<<<(8 * 4096 + 255) / 256, 256, 0, stream>>>(node_w1, nw1p, 2 * HD, 8);
    convert_w_frag_kernel<<<(4 * 4096 + 255) / 256, 256, 0, stream>>>(node_w2, nw2p, HD, 4);

    // edge phase: message MLP + coord head + scatter-adds
    long long etiles = (Ee + 15) / 16;
    unsigned eblocks = (unsigned)((etiles + EW - 1) / EW);
    egnn_edge_kernel<<<eblocks, EW * 32, 0, stream>>>(
        h, x, ei, w1p, msg_b1, w2p, msg_b2, cw1p, coord_b1, coord_w2,
        agg, coordacc, deg, Ee);

    // node phase: node MLP + residual + LayerNorm -> h_new
    int ntiles = (Nn + 15) / 16;
    unsigned nblocks = (unsigned)((ntiles + NW - 1) / NW);
    egnn_node_kernel<<<nblocks, NW * 32, 0, stream>>>(
        h, agg, nw1p, node_b1, nw2p, node_b2, ln_g, ln_b, (float*)d_out, Nn);

    // x_new
    xnew_kernel<<<(Nn + 255) / 256, 256, 0, stream>>>(
        x, coordacc, deg, (float*)d_out + (size_t)Nn * HD, Nn);
}
